// Grid_3264175145671
// MI455X (gfx1250) — matrix operand order repaired and checked by reading, not compiled, not run on hardware
//
#include <hip/hip_runtime.h>
#include <hip/hip_bf16.h>
#include <stdint.h>

// ---- CDNA5 WMMA vector types (wave32) ----
typedef _Float16 v16h __attribute__((ext_vector_type(16)));
typedef float    v8f  __attribute__((ext_vector_type(8)));

#define LNUM   16
#define FNUM   2
#define TSIZE  (1u << 19)
#define OUTRES 512
#define WPB    8            // waves per block
#define P1C    2654435761u
#define P2C    805459861u

union PackH2 { unsigned int u; _Float16 h[2]; };
static __device__ __forceinline__ unsigned int packHalf2(float a, float b) {
    PackH2 p; p.h[0] = (_Float16)a; p.h[1] = (_Float16)b; return p.u;
}

struct I4x2 { int4 a, b; };

// 16 contiguous f16 from LDS -> v16h (two ds_load_b128)
static __device__ __forceinline__ v16h ldsFrag16(const _Float16* p) {
    I4x2 t;
    t.a = *reinterpret_cast<const int4*>(p);
    t.b = *reinterpret_cast<const int4*>(p + 8);
    return __builtin_bit_cast(v16h, t);
}

// A-fragment (16x32 f16 WMMA layout) from a row-major [16][64] f16 LDS tile.
// lanes 0-15: K 0..15 ; lanes 16-31: K 16..31 (offset kbase)
static __device__ __forceinline__ v16h fragA64(const _Float16* h, int row, int hi, int kbase) {
    const _Float16* p = h + row * 64 + kbase + hi * 16;
    I4x2 t;
    t.a = *reinterpret_cast<const int4*>(p);
    t.b = *reinterpret_cast<const int4*>(p + 8);
    return __builtin_bit_cast(v16h, t);
}

__global__ __launch_bounds__(256) void ngp_fused_kernel(
    const float* __restrict__ xyz,
    const float* __restrict__ table,
    const float* __restrict__ W0,
    const float* __restrict__ W1,
    const float* __restrict__ W2,
    const int*   __restrict__ boundPtr,
    float*       __restrict__ out,
    int nPts, int tilesTotal, int tilesPerWave)
{
    // Column-major f16 weight copies (contiguous K per output column -> B frags)
    __shared__ __align__(16) _Float16 w0cm[64 * 32];   // [col][k]   4 KB
    __shared__ __align__(16) _Float16 w1cm[64 * 64];   // [col][k]   8 KB
    __shared__ __align__(16) _Float16 w2cm[16 * 64];   // [col][k]   2 KB (cols 8..15 zero)
    // Per-wave scratch, reused in sequence: feature tile (as f16 pairs) -> H0 -> H1.
    // Safe because A-fragments are pulled into registers before the next stage's
    // stores, and same-wave DS ops are processed in order.
    __shared__ __align__(16) _Float16 scr[WPB][16 * 64];       // 2 KB/wave = 16 KB

    const int tid = threadIdx.x;

    for (int i = tid; i < 64 * 32; i += 256) {
        int n = i >> 5, k = i & 31;
        w0cm[n * 32 + k] = (_Float16)W0[k * 64 + n];
    }
    for (int i = tid; i < 64 * 64; i += 256) {
        int n = i >> 6, k = i & 63;
        w1cm[n * 64 + k] = (_Float16)W1[k * 64 + n];
    }
    for (int i = tid; i < 16 * 64; i += 256) {
        int n = i >> 6, k = i & 63;
        w2cm[n * 64 + k] = (n < 8) ? (_Float16)W2[k * 8 + n] : (_Float16)0.0f;
    }
    __syncthreads();

    const int wave = tid >> 5;
    const int lane = tid & 31;
    const int hi   = lane >> 4;     // which half of the wave
    const int ln   = lane & 15;

    const float bf    = (float)boundPtr[0];
    const float inv2b = 1.0f / (2.0f * bf);

    _Float16*     sw = scr[wave];
    unsigned int* fp = reinterpret_cast<unsigned int*>(sw);   // feature-pair view

    const int gw = blockIdx.x * WPB + wave;
    const int tBeg = gw * tilesPerWave;
    int tEnd = tBeg + tilesPerWave;
    if (tEnd > tilesTotal) tEnd = tilesTotal;

    for (int t = tBeg; t < tEnd; ++t) {
        // ---------- encoding: row = ln, levels split even/odd by lane half ----------
        {
            const int row = ln;
            const int pt  = t * 2 + (row >> 3);
            const int bc  = row & 7;
            float x = 0.f, y = 0.f, z = 0.f;
            if (pt < nPts) { x = xyz[pt * 3 + 0]; y = xyz[pt * 3 + 1]; z = xyz[pt * 3 + 2]; }
            const float cxf = (x + bf) * inv2b * (float)OUTRES;
            const float cyf = (y + bf) * inv2b * (float)OUTRES;
            const float czf = (z + bf) * inv2b * (float)OUTRES;
            const float c0x = fminf(fmaxf(floorf(cxf), 0.f), 511.f);
            const float c0y = fminf(fmaxf(floorf(cyf), 0.f), 511.f);
            const float c0z = fminf(fmaxf(floorf(czf), 0.f), 511.f);
            const unsigned int qx = (unsigned int)c0x + (unsigned int)(bc & 1);
            const unsigned int qy = (unsigned int)c0y + (unsigned int)((bc >> 1) & 1);
            const unsigned int qz = (unsigned int)c0z + (unsigned int)((bc >> 2) & 1);

            #pragma unroll
            for (int j = 0; j < 8; ++j) {
                const int l = j * 2 + hi;
                const float* tb = table + (size_t)l * (size_t)(TSIZE * FNUM);
                float f0, f1;
                if (l < 5) {
                    // sample not on this level's lattice: full 8-corner trilerp
                    const int s = 5 - l;
                    const unsigned int m = (1u << s) - 1u;
                    const unsigned int px = qx >> s, py = qy >> s, pz = qz >> s;
                    const float inv = 1.0f / (float)(1 << s);
                    const float fx = (float)(qx & m) * inv;
                    const float fy = (float)(qy & m) * inv;
                    const float fz = (float)(qz & m) * inv;
                    const unsigned int res = 16u << l;
                    f0 = 0.f; f1 = 0.f;
                    #pragma unroll
                    for (int c = 0; c < 8; ++c) {
                        unsigned int cx = px + (unsigned int)(c & 1);
                        unsigned int cy = py + (unsigned int)((c >> 1) & 1);
                        unsigned int cz = pz + (unsigned int)((c >> 2) & 1);
                        unsigned int idx;
                        if (l < 3) {                       // dense levels 0..2
                            cx = cx < res ? cx : res;
                            cy = cy < res ? cy : res;
                            cz = cz < res ? cz : res;
                            idx = cx + cy * (res + 1u) + cz * (res + 1u) * (res + 1u);
                        } else {                           // hashed levels 3..4
                            idx = (cx ^ (cy * P1C) ^ (cz * P2C)) & (TSIZE - 1u);
                        }
                        const float2 g = *(const float2*)(tb + (size_t)idx * 2u);
                        const float w = ((c & 1) ? fx : 1.f - fx) *
                                        ((c & 2) ? fy : 1.f - fy) *
                                        ((c & 4) ? fz : 1.f - fz);
                        f0 += w * g.x; f1 += w * g.y;
                    }
                } else {
                    // res multiple of 512: exact lattice point -> single gather
                    const unsigned int sh = (unsigned int)(l - 5);
                    const unsigned int cx = qx << sh, cy = qy << sh, cz = qz << sh;
                    const unsigned int idx = (cx ^ (cy * P1C) ^ (cz * P2C)) & (TSIZE - 1u);
                    const float2 g = *(const float2*)(tb + (size_t)idx * 2u);
                    f0 = g.x; f1 = g.y;
                }
                fp[row * 16 + l] = packHalf2(f0, f1);
            }
        }
        __builtin_amdgcn_wave_barrier();

        // ---------- A fragment (16x32 f16) from feature tile ----------
        const unsigned int* ab = &fp[ln * 16 + (hi ? 8 : 0)];
        v16h A;
        {
            I4x2 tA;
            tA.a = *reinterpret_cast<const int4*>(ab);
            tA.b = *reinterpret_cast<const int4*>(ab + 4);
            A = __builtin_bit_cast(v16h, tA);
        }
        __builtin_amdgcn_wave_barrier();

        // ---------- layer 1: [16x32] @ [32x64] + ReLU -> scr (H0) ----------
        #pragma unroll
        for (int c = 0; c < 4; ++c) {
            v16h B = ldsFrag16(&w0cm[(c * 16 + ln) * 32 + hi * 16]);
            v8f zacc = {};
            v8f acc = __builtin_amdgcn_wmma_f32_16x16x32_f16(
                false, A, false, B, (short)0, zacc, false, false);
            #pragma unroll
            for (int r = 0; r < 8; ++r) {
                sw[(r + 8 * hi) * 64 + c * 16 + ln] = (_Float16)fmaxf(acc[r], 0.f);
            }
        }
        __builtin_amdgcn_wave_barrier();

        // ---------- layer 2: [16x64] @ [64x64] + ReLU -> scr (H1) ----------
        v16h A20 = fragA64(sw, ln, hi, 0);
        v16h A21 = fragA64(sw, ln, hi, 32);
        __builtin_amdgcn_wave_barrier();
        #pragma unroll
        for (int c = 0; c < 4; ++c) {
            v16h B0 = ldsFrag16(&w1cm[(c * 16 + ln) * 64 + 0  + hi * 16]);
            v16h B1 = ldsFrag16(&w1cm[(c * 16 + ln) * 64 + 32 + hi * 16]);
            v8f zacc = {};
            v8f acc = __builtin_amdgcn_wmma_f32_16x16x32_f16(
                false, A20, false, B0, (short)0, zacc, false, false);
            acc = __builtin_amdgcn_wmma_f32_16x16x32_f16(
                false, A21, false, B1, (short)0, acc, false, false);
            #pragma unroll
            for (int r = 0; r < 8; ++r) {
                sw[(r + 8 * hi) * 64 + c * 16 + ln] = (_Float16)fmaxf(acc[r], 0.f);
            }
        }
        __builtin_amdgcn_wave_barrier();

        // ---------- layer 3: [16x64] @ [64x16(pad)] ----------
        v16h A30 = fragA64(sw, ln, hi, 0);
        v16h A31 = fragA64(sw, ln, hi, 32);
        v16h B30 = ldsFrag16(&w2cm[ln * 64 + 0  + hi * 16]);
        v16h B31 = ldsFrag16(&w2cm[ln * 64 + 32 + hi * 16]);
        v8f zacc = {};
        v8f F = __builtin_amdgcn_wmma_f32_16x16x32_f16(
            false, A30, false, B30, (short)0, zacc, false, false);
        F = __builtin_amdgcn_wmma_f32_16x16x32_f16(
            false, A31, false, B31, (short)0, F, false, false);

        // ---------- trilinear blend over the 8 outer corners (in-register) ----------
        // D layout: VGPR r, this lane -> row (r + 8*hi) = point (2t+hi), corner r,
        // output channel = ln. All 8 corners of one point live in this lane's 8 accs.
        if (ln < 8) {
            const int pt = t * 2 + hi;
            if (pt < nPts) {
                const float x = xyz[pt * 3 + 0];
                const float y = xyz[pt * 3 + 1];
                const float z = xyz[pt * 3 + 2];
                const float cxf = (x + bf) * inv2b * (float)OUTRES;
                const float cyf = (y + bf) * inv2b * (float)OUTRES;
                const float czf = (z + bf) * inv2b * (float)OUTRES;
                const float fx = cxf - fminf(fmaxf(floorf(cxf), 0.f), 511.f);
                const float fy = cyf - fminf(fmaxf(floorf(cyf), 0.f), 511.f);
                const float fz = czf - fminf(fmaxf(floorf(czf), 0.f), 511.f);
                float sum = 0.f;
                #pragma unroll
                for (int r = 0; r < 8; ++r) {
                    const float w = ((r & 1) ? fx : 1.f - fx) *
                                    ((r & 2) ? fy : 1.f - fy) *
                                    ((r & 4) ? fz : 1.f - fz);
                    sum += w * F[r];
                }
                out[pt * 8 + ln] = sum;
            }
        }
        __builtin_amdgcn_wave_barrier();
    }
}

extern "C" void kernel_launch(void* const* d_in, const int* in_sizes, int n_in,
                              void* d_out, int out_size, void* d_ws, size_t ws_size,
                              hipStream_t stream) {
    const float* xyz   = (const float*)d_in[0];
    const float* table = (const float*)d_in[1];
    const float* W0    = (const float*)d_in[2];
    const float* W1    = (const float*)d_in[3];
    const float* W2    = (const float*)d_in[4];
    const int*   bound = (const int*)d_in[5];
    float* out = (float*)d_out;

    const int nPts  = in_sizes[0] / 3;
    const int tiles = (nPts * 8 + 15) / 16;      // 16 rows (= 2 points x 8 corners) per tile

    const int blocks     = 2048;                 // 8 waves each -> 16384 waves
    const int wavesTotal = blocks * WPB;
    const int tilesPerWave = (tiles + wavesTotal - 1) / wavesTotal;

    ngp_fused_kernel<<<blocks, 256, 0, stream>>>(
        xyz, table, W0, W1, W2, bound, out, nPts, tiles, tilesPerWave);
}
